// Seq2Seq_47725676593566
// MI455X (gfx1250) — matrix operand order repaired and checked by reading, not compiled, not run on hardware
//
#include <hip/hip_runtime.h>
#include <cstdint>
#include <cstddef>

// ---------------- problem dims ----------------
#define B_    32
#define S_    96
#define T_    64
#define E_    256
#define HE_   512
#define HD_   512
#define ENCD_ 1024      // 2*HE
#define DECIN_ 1280     // E + ENCD
#define VT_   16000
#define G4_   2048      // 4*HE == 4*HD

typedef __attribute__((ext_vector_type(16))) _Float16 v16h;
typedef __attribute__((ext_vector_type(8)))  float    v8f;

union Frag { v16h v; uint32_t u[8]; };

__device__ __forceinline__ float sigf(float x) { return 1.0f / (1.0f + expf(-x)); }

// ============================================================
// Generic WMMA GEMM:  C[M,N] = act( A[M,K] @ W[N,K]^T + bias )
// A,W f16 row-major; C f32 (ldc), optional f16 mirror Cbf.
// One wave computes NT consecutive 16x16 C tiles along N, sharing
// one A fragment per K-chunk (4 waves per block along N-groups).
// Requires N % (16*NT) == 0.
// ============================================================
template <int NT>
__global__ __launch_bounds__(128) void k_gemm(
    const _Float16* __restrict__ A, int lda,
    const _Float16* __restrict__ W,          // [N,K] row-major
    const float* __restrict__ bias,          // nullable, per-N
    float* __restrict__ C, long long ldc,
    _Float16* __restrict__ Cbf, int ldcb,    // nullable
    int M, int N, int K, int act)            // act: 0=none 1=tanh
{
  const int lane = threadIdx.x & 31;
  const int wv   = threadIdx.x >> 5;
  const int g    = blockIdx.x * 4 + wv;      // N tile-group index
  const int n0   = g * 16 * NT;
  if (n0 >= N) return;
  const int m0 = blockIdx.y * 16;
  const int hl = lane >> 4, l16 = lane & 15;

  const _Float16* Ar = A + (long long)(m0 + l16) * lda;
  const _Float16* Wr[NT];
#pragma unroll
  for (int t = 0; t < NT; ++t)
    Wr[t] = W + (long long)(n0 + t * 16 + l16) * K;

  v8f acc[NT];
#pragma unroll
  for (int t = 0; t < NT; ++t) acc[t] = (v8f){};

  for (int k0 = 0; k0 < K; k0 += 32) {
    if (k0 + 32 < K) {
#pragma unroll
      for (int t = 0; t < NT; ++t)
        __builtin_prefetch((const void*)(Wr[t] + k0 + 32), 0, 3);
    }
    Frag a;
#pragma unroll
    for (int i = 0; i < 8; ++i) {
      // ISA 16-bit A 16x32 layout: lanes<16 K in {0..7,16..23}, lanes>=16 +8
      int ka = ((i < 4) ? 2 * i : 16 + 2 * (i - 4)) + 8 * hl;
      a.u[i] = *(const uint32_t*)(Ar + k0 + ka);
    }
#pragma unroll
    for (int t = 0; t < NT; ++t) {
      Frag b;
#pragma unroll
      for (int i = 0; i < 8; ++i) {
        // B 32x16 layout matches A: lanes<16 K in {0..7,16..23}, lanes>=16 +8
        int kb = ((i < 4) ? 2 * i : 16 + 2 * (i - 4)) + 8 * hl;
        b.u[i] = *(const uint32_t*)(Wr[t] + k0 + kb);
      }
      acc[t] = __builtin_amdgcn_wmma_f32_16x16x32_f16(false, a.v, false, b.v,
                                                      (short)0, acc[t], false, false);
    }
  }

#pragma unroll
  for (int t = 0; t < NT; ++t) {
    int nc = n0 + t * 16 + l16;
    float bs = bias ? bias[nc] : 0.0f;
#pragma unroll
    for (int i = 0; i < 8; ++i) {
      float r = acc[t][i] + bs;
      if (act == 1) r = tanhf(r);
      long long row = m0 + i + 8 * hl;      // C layout: VGPR i -> M=i (+8 hi lanes)
      C[row * ldc + nc] = r;
      if (Cbf) Cbf[row * (long long)ldcb + nc] = (_Float16)r;
    }
  }
}

// ---------------- utility kernels ----------------
__global__ void k_f2h(_Float16* d, const float* s, int n) {
  int i = blockIdx.x * blockDim.x + threadIdx.x;
  if (i < n) d[i] = (_Float16)s[i];
}
__global__ void k_zero_h(_Float16* d, int n) {
  int i = blockIdx.x * blockDim.x + threadIdx.x;
  if (i < n) d[i] = (_Float16)0.0f;
}
__global__ void k_zero_f(float* d, int n) {
  int i = blockIdx.x * blockDim.x + threadIdx.x;
  if (i < n) d[i] = 0.0f;
}
// dst[rows, na+nb] = [A | B], f32 sources -> f16
__global__ void k_concat2_f2h(_Float16* d, const float* A, int na,
                              const float* Bp, int nb, int rows) {
  int n = na + nb;
  int i = blockIdx.x * blockDim.x + threadIdx.x;
  if (i >= rows * n) return;
  int r = i / n, j = i % n;
  float v = (j < na) ? A[(long long)r * na + j] : Bp[(long long)r * nb + (j - na)];
  d[i] = (_Float16)v;
}
// f16 sources
__global__ void k_concat2_h(_Float16* d, const _Float16* A, int na,
                            const _Float16* Bp, int nb, int rows) {
  int n = na + nb;
  int i = blockIdx.x * blockDim.x + threadIdx.x;
  if (i >= rows * n) return;
  int r = i / n, j = i % n;
  d[i] = (j < na) ? A[(long long)r * na + j] : Bp[(long long)r * nb + (j - na)];
}
// embedding gather -> f16
__global__ void k_embed(_Float16* o, const float* tab, const int* ids, int ntok, int E) {
  int i = blockIdx.x * blockDim.x + threadIdx.x;
  if (i >= ntok * E) return;
  int t = i / E, e = i % E;
  o[i] = (_Float16)tab[(long long)ids[t] * E + e];
}
__global__ void k_embed_dec(_Float16* o, const float* tab, const int* tg) {
  int i = blockIdx.x * blockDim.x + threadIdx.x;
  if (i >= B_ * (T_ - 1) * E_) return;
  int b = i / ((T_ - 1) * E_);
  int r = i % ((T_ - 1) * E_);
  int t = r / E_, e = r % E_;
  o[i] = (_Float16)tab[(long long)tg[b * T_ + t] * E_ + e];
}

// xcat0 = [emb_t(256) | ctx(1024) | h0(512)]  -> [32,1792]
__global__ void k_build_xcat0(_Float16* x, const _Float16* embdec, int t,
                              const _Float16* ctx, const _Float16* h0) {
  int i = blockIdx.x * blockDim.x + threadIdx.x;
  if (i >= B_ * (E_ + ENCD_ + HD_)) return;
  int nb = E_ + ENCD_ + HD_;                    // 1792
  int b = i / nb, j = i % nb;
  _Float16 v;
  if (j < E_)              v = embdec[((long long)b * (T_ - 1) + t) * E_ + j];
  else if (j < E_ + ENCD_) v = ctx[(long long)b * ENCD_ + (j - E_)];
  else                     v = h0[(long long)b * HD_ + (j - E_ - ENCD_)];
  x[i] = v;
}

// ---------------- LSTM cells ----------------
// encoder: gates = gx(t) + gh ; masked update ; writes h (f16 slice) + out slice
__global__ void k_enc_cell(const float* __restrict__ gx, const float* __restrict__ gh,
                           float* __restrict__ c, _Float16* __restrict__ hbf, int ldh,
                           _Float16* __restrict__ outp, const int* __restrict__ lens, int t) {
  int i = blockIdx.x * blockDim.x + threadIdx.x;
  if (i >= B_ * HE_) return;
  int b = i >> 9, j = i & 511;
  const float* gr = gx + (long long)(b * S_ + t) * G4_;
  const float* hr = gh + (long long)b * G4_;
  float gi = gr[j]           + hr[j];
  float gf = gr[HE_ + j]     + hr[HE_ + j];
  float gg = gr[2 * HE_ + j] + hr[2 * HE_ + j];
  float go = gr[3 * HE_ + j] + hr[3 * HE_ + j];
  float cn = sigf(gf) * c[i] + sigf(gi) * tanhf(gg);
  float hn = sigf(go) * tanhf(cn);
  bool valid = t < lens[b];
  if (valid) { c[i] = cn; hbf[(long long)b * ldh + j] = (_Float16)hn; }
  outp[(long long)(b * S_ + t) * ENCD_ + j] = (_Float16)(valid ? hn : 0.0f);
}

// decoder: no mask, gates already summed (concat GEMM)
__global__ void k_dec_cell(const float* __restrict__ g, float* __restrict__ c,
                           _Float16* __restrict__ hbf) {
  int i = blockIdx.x * blockDim.x + threadIdx.x;
  if (i >= B_ * HD_) return;
  int b = i >> 9, j = i & 511;
  const float* gr = g + (long long)b * G4_;
  float cn = sigf(gr[HD_ + j]) * c[i] + sigf(gr[j]) * tanhf(gr[2 * HD_ + j]);
  float hn = sigf(gr[3 * HD_ + j]) * tanhf(cn);
  c[i] = cn;
  hbf[i] = (_Float16)hn;
}

// ---------------- attention ----------------
// scores[b,s] = enc_out[b,s,:] . wdh[b,:]  (masked with -inf where src==0)
__global__ __launch_bounds__(128) void k_attn_scores(const _Float16* __restrict__ eo,
                                                     const float* __restrict__ wdh,
                                                     const int* __restrict__ src,
                                                     float* __restrict__ sc) {
  int gw = blockIdx.x * 4 + (threadIdx.x >> 5);
  int lane = threadIdx.x & 31;
  if (gw >= B_ * S_) return;
  int b = gw / S_, s = gw % S_;
  const _Float16* er = eo + (long long)(b * S_ + s) * ENCD_;
  const float* wr = wdh + (long long)b * ENCD_;
  float acc = 0.0f;
  for (int e = lane; e < ENCD_; e += 32) acc += (float)er[e] * wr[e];
  for (int off = 16; off; off >>= 1) acc += __shfl_down(acc, off);
  if (lane == 0) sc[gw] = (src[b * S_ + s] == 0) ? -__builtin_inff() : acc;
}

__global__ __launch_bounds__(128) void k_softmax96(const float* __restrict__ sc,
                                                   float* __restrict__ a) {
  __shared__ float red[128];
  int b = blockIdx.x, s = threadIdx.x;
  float v = (s < S_) ? sc[b * S_ + s] : -__builtin_inff();
  red[s] = v; __syncthreads();
  for (int off = 64; off; off >>= 1) { if (s < off) red[s] = fmaxf(red[s], red[s + off]); __syncthreads(); }
  float mx = red[0]; __syncthreads();
  float e = (s < S_) ? expf(v - mx) : 0.0f;
  red[s] = e; __syncthreads();
  for (int off = 64; off; off >>= 1) { if (s < off) red[s] += red[s + off]; __syncthreads(); }
  float sum = red[0];
  if (s < S_) a[b * S_ + s] = e / sum;
}

__global__ void k_attn_ctx(const _Float16* __restrict__ eo, const float* __restrict__ a,
                           _Float16* __restrict__ ctx) {
  int i = blockIdx.x * blockDim.x + threadIdx.x;
  if (i >= B_ * ENCD_) return;
  int b = i >> 10, e = i & 1023;
  float acc = 0.0f;
  for (int s = 0; s < S_; ++s)
    acc += a[b * S_ + s] * (float)eo[(long long)(b * S_ + s) * ENCD_ + e];
  ctx[i] = (_Float16)acc;
}

// decoder initial state: layer0 <- bridge[0], layers1..3 <- bridge[1]
__global__ void k_init_dec(_Float16* hdec, float* cdec,
                           const _Float16* bh, const float* cb) {
  int i = blockIdx.x * blockDim.x + threadIdx.x;
  if (i >= 4 * B_ * HD_) return;
  int l = i >> 14, r = i & 16383;
  int sl = l ? 1 : 0;
  hdec[i] = bh[sl * B_ * HD_ + r];
  cdec[i] = cb[sl * B_ * HD_ + r];
}

// ---------------- host-side gemm launcher ----------------
static inline void gemm(hipStream_t st, const _Float16* A, int lda, const _Float16* W,
                        const float* bias, float* C, long long ldc,
                        _Float16* Cbf, int ldcb, int M, int N, int K, int act,
                        int nt /*1 or 4*/) {
  if (nt == 4) {
    dim3 g((N / 64 + 3) / 4, M / 16);
    k_gemm<4><<<g, 128, 0, st>>>(A, lda, W, bias, C, ldc, Cbf, ldcb, M, N, K, act);
  } else {
    dim3 g((N / 16 + 3) / 4, M / 16);
    k_gemm<1><<<g, 128, 0, st>>>(A, lda, W, bias, C, ldc, Cbf, ldcb, M, N, K, act);
  }
}
static inline int gr(int n) { return (n + 255) / 256; }

extern "C" void kernel_launch(void* const* d_in, const int* in_sizes, int n_in,
                              void* d_out, int out_size, void* d_ws, size_t ws_size,
                              hipStream_t stream) {
  const int*   src     = (const int*)d_in[0];
  const int*   lens    = (const int*)d_in[1];
  const int*   tgt     = (const int*)d_in[2];
  const float* enc_emb = (const float*)d_in[3];
  const float* eWih0   = (const float*)d_in[4];
  const float* eWhh0   = (const float*)d_in[5];
  const float* eb0     = (const float*)d_in[6];
  const float* eWih1   = (const float*)d_in[7];
  const float* eWhh1   = (const float*)d_in[8];
  const float* eb1     = (const float*)d_in[9];
  const float* brhW    = (const float*)d_in[10];
  const float* brhb    = (const float*)d_in[11];
  const float* brcW    = (const float*)d_in[12];
  const float* brcb    = (const float*)d_in[13];
  const float* dec_emb = (const float*)d_in[14];
  const float* dWih0   = (const float*)d_in[15];
  const float* dWhh0   = (const float*)d_in[16];
  const float* db0     = (const float*)d_in[17];
  const float* dWihL   = (const float*)d_in[18];
  const float* dWhhL   = (const float*)d_in[19];
  const float* dbL     = (const float*)d_in[20];
  const float* attnW   = (const float*)d_in[21];
  const float* projW   = (const float*)d_in[22];
  const float* projb   = (const float*)d_in[23];
  float* outp = (float*)d_out;

  // ---- bump allocator over d_ws ----
  char* wsb = (char*)d_ws; size_t off = 0;
  auto alloc = [&](size_t bytes) -> void* {
    void* p = wsb + off; off = (off + bytes + 255) & ~(size_t)255; return p; };

  // f16 weights
  _Float16* wih0  = (_Float16*)alloc((size_t)2 * G4_ * E_ * 2);
  _Float16* whh0  = (_Float16*)alloc((size_t)2 * G4_ * HE_ * 2);
  _Float16* wih1  = (_Float16*)alloc((size_t)2 * G4_ * ENCD_ * 2);
  _Float16* whh1  = (_Float16*)alloc((size_t)2 * G4_ * HE_ * 2);
  _Float16* w0cat = (_Float16*)alloc((size_t)G4_ * (DECIN_ + HD_) * 2);   // [2048,1792]
  _Float16* wlcat = (_Float16*)alloc((size_t)3 * G4_ * (2 * HD_) * 2);    // 3x[2048,1024]
  _Float16* attnw = (_Float16*)alloc((size_t)ENCD_ * HD_ * 2);
  _Float16* projw = (_Float16*)alloc((size_t)VT_ * (HD_ + ENCD_) * 2);
  _Float16* brhw  = (_Float16*)alloc((size_t)HD_ * ENCD_ * 2);
  _Float16* brcw  = (_Float16*)alloc((size_t)HD_ * ENCD_ * 2);
  // activations
  _Float16* x0     = (_Float16*)alloc((size_t)B_ * S_ * E_ * 2);
  _Float16* x1     = (_Float16*)alloc((size_t)B_ * S_ * ENCD_ * 2);
  _Float16* encout = (_Float16*)alloc((size_t)B_ * S_ * ENCD_ * 2);
  _Float16* embdec = (_Float16*)alloc((size_t)B_ * (T_ - 1) * E_ * 2);
  float*    gatesx = (float*)alloc((size_t)B_ * S_ * G4_ * 4);
  float*    gatesh = (float*)alloc((size_t)B_ * G4_ * 4);
  _Float16* hcat   = (_Float16*)alloc((size_t)2 * B_ * ENCD_ * 2);  // [layer][32,1024] f/b halves
  float*    cenc   = (float*)alloc((size_t)2 * 2 * B_ * HE_ * 4);   // [layer][dir][32,512]
  _Float16* ccat   = (_Float16*)alloc((size_t)2 * B_ * ENCD_ * 2);
  float*    bhf    = (float*)alloc((size_t)2 * B_ * HD_ * 4);
  _Float16* bhbf   = (_Float16*)alloc((size_t)2 * B_ * HD_ * 2);
  float*    cbf    = (float*)alloc((size_t)2 * B_ * HD_ * 4);
  _Float16* hdec   = (_Float16*)alloc((size_t)4 * B_ * HD_ * 2);
  float*    cdec   = (float*)alloc((size_t)4 * B_ * HD_ * 4);
  float*    wdh    = (float*)alloc((size_t)B_ * ENCD_ * 4);
  float*    scores = (float*)alloc((size_t)B_ * S_ * 4);
  float*    attp   = (float*)alloc((size_t)B_ * S_ * 4);
  _Float16* ctxh   = (_Float16*)alloc((size_t)B_ * ENCD_ * 2);
  _Float16* xcat0  = (_Float16*)alloc((size_t)B_ * (E_ + ENCD_ + HD_) * 2);
  _Float16* xcatl  = (_Float16*)alloc((size_t)B_ * (2 * HD_) * 2);
  _Float16* pcat   = (_Float16*)alloc((size_t)B_ * (HD_ + ENCD_) * 2);
  float*    gdec   = (float*)alloc((size_t)B_ * G4_ * 4);

  // ---- weight conversion ----
  auto f2h = [&](_Float16* d, const float* s, int n) {
    k_f2h<<<gr(n), 256, 0, stream>>>(d, s, n); };
  f2h(wih0, eWih0, 2 * G4_ * E_);
  f2h(whh0, eWhh0, 2 * G4_ * HE_);
  f2h(wih1, eWih1, 2 * G4_ * ENCD_);
  f2h(whh1, eWhh1, 2 * G4_ * HE_);
  f2h(attnw, attnW, ENCD_ * HD_);
  f2h(projw, projW, VT_ * (HD_ + ENCD_));
  f2h(brhw, brhW, HD_ * ENCD_);
  f2h(brcw, brcW, HD_ * ENCD_);
  k_concat2_f2h<<<gr(G4_ * (DECIN_ + HD_)), 256, 0, stream>>>(
      w0cat, dWih0, DECIN_, dWhh0, HD_, G4_);
  for (int l = 0; l < 3; ++l)
    k_concat2_f2h<<<gr(G4_ * 2 * HD_), 256, 0, stream>>>(
        wlcat + (size_t)l * G4_ * 2 * HD_,
        dWihL + (size_t)l * G4_ * HD_, HD_,
        dWhhL + (size_t)l * G4_ * HD_, HD_, G4_);

  // ---- embeddings ----
  k_embed<<<gr(B_ * S_ * E_), 256, 0, stream>>>(x0, enc_emb, src, B_ * S_, E_);
  k_embed_dec<<<gr(B_ * (T_ - 1) * E_), 256, 0, stream>>>(embdec, dec_emb, tgt);

  // ---- zero recurrent state ----
  k_zero_h<<<gr(2 * B_ * ENCD_), 256, 0, stream>>>(hcat, 2 * B_ * ENCD_);
  k_zero_f<<<gr(2 * 2 * B_ * HE_), 256, 0, stream>>>(cenc, 2 * 2 * B_ * HE_);

  // ---- encoder ----
  const float* ebias[2] = { eb0, eb1 };
  for (int layer = 0; layer < 2; ++layer) {
    const _Float16* wih = layer ? wih1 : wih0;
    const _Float16* whh = layer ? whh1 : whh0;
    const _Float16* xin = layer ? x1 : x0;
    int kin = layer ? ENCD_ : E_;
    _Float16* outdst = layer ? encout : x1;
    for (int dir = 0; dir < 2; ++dir) {
      const _Float16* wihd = wih + (size_t)dir * G4_ * kin;
      const _Float16* whhd = whh + (size_t)dir * G4_ * HE_;
      const float*    bd   = ebias[layer] + (size_t)dir * G4_;
      // time-batched input projection + bias (M=3072: NT=4, plenty of waves)
      gemm(stream, xin, kin, wihd, bd, gatesx, G4_, nullptr, 0,
           B_ * S_, G4_, kin, 0, 4);
      _Float16* hsl = hcat + (size_t)layer * B_ * ENCD_ + dir * HE_;  // strided slice
      float*    csl = cenc + ((size_t)layer * 2 + dir) * B_ * HE_;
      _Float16* osl = outdst + dir * HE_;
      for (int step = 0; step < S_; ++step) {
        int t = dir ? (S_ - 1 - step) : step;
        gemm(stream, hsl, ENCD_, whhd, nullptr, gatesh, G4_, nullptr, 0,
             B_, G4_, HE_, 0, 1);
        k_enc_cell<<<gr(B_ * HE_), 256, 0, stream>>>(
            gatesx, gatesh, csl, hsl, ENCD_, osl, lens, t);
      }
    }
  }

  // ---- bridge ----
  for (int l = 0; l < 2; ++l)
    k_concat2_f2h<<<gr(B_ * ENCD_), 256, 0, stream>>>(
        ccat + (size_t)l * B_ * ENCD_,
        cenc + ((size_t)l * 2 + 0) * B_ * HE_, HE_,
        cenc + ((size_t)l * 2 + 1) * B_ * HE_, HE_, B_);
  for (int l = 0; l < 2; ++l) {
    gemm(stream, hcat + (size_t)l * B_ * ENCD_, ENCD_, brhw, brhb,
         bhf + (size_t)l * B_ * HD_, HD_, bhbf + (size_t)l * B_ * HD_, HD_,
         B_, HD_, ENCD_, 1, 1);
    gemm(stream, ccat + (size_t)l * B_ * ENCD_, ENCD_, brcw, brcb,
         cbf + (size_t)l * B_ * HD_, HD_, nullptr, 0,
         B_, HD_, ENCD_, 1, 1);
  }
  k_init_dec<<<gr(4 * B_ * HD_), 256, 0, stream>>>(hdec, cdec, bhbf, cbf);

  // ---- decoder ----
  for (int t = 0; t < T_ - 1; ++t) {
    // attention (uses h[3] from previous step / bridge)
    gemm(stream, hdec + (size_t)3 * B_ * HD_, HD_, attnw, nullptr,
         wdh, ENCD_, nullptr, 0, B_, ENCD_, HD_, 0, 1);
    k_attn_scores<<<(B_ * S_ + 3) / 4, 128, 0, stream>>>(encout, wdh, src, scores);
    k_softmax96<<<B_, 128, 0, stream>>>(scores, attp);
    k_attn_ctx<<<gr(B_ * ENCD_), 256, 0, stream>>>(encout, attp, ctxh);
    // layer 0: x = [emb_t | ctx | h0]  with W = [Wih0 | Whh0]
    k_build_xcat0<<<gr(B_ * (E_ + ENCD_ + HD_)), 256, 0, stream>>>(
        xcat0, embdec, t, ctxh, hdec);
    gemm(stream, xcat0, E_ + ENCD_ + HD_, w0cat, db0, gdec, G4_, nullptr, 0,
         B_, G4_, E_ + ENCD_ + HD_, 0, 1);
    k_dec_cell<<<gr(B_ * HD_), 256, 0, stream>>>(gdec, cdec, hdec);
    // layers 1..3
    for (int l = 1; l < 4; ++l) {
      k_concat2_h<<<gr(B_ * 2 * HD_), 256, 0, stream>>>(
          xcatl, hdec + (size_t)(l - 1) * B_ * HD_, HD_,
          hdec + (size_t)l * B_ * HD_, HD_, B_);
      gemm(stream, xcatl, 2 * HD_, wlcat + (size_t)(l - 1) * G4_ * 2 * HD_,
           dbL + (size_t)(l - 1) * G4_, gdec, G4_, nullptr, 0,
           B_, G4_, 2 * HD_, 0, 1);
      k_dec_cell<<<gr(B_ * HD_), 256, 0, stream>>>(
          gdec, cdec + (size_t)l * B_ * HD_, hdec + (size_t)l * B_ * HD_);
    }
    // projection: [h3 | ctx] @ projW^T + b -> out[:, t, :]  (NT=4: A-fragment reuse)
    k_concat2_h<<<gr(B_ * (HD_ + ENCD_)), 256, 0, stream>>>(
        pcat, hdec + (size_t)3 * B_ * HD_, HD_, ctxh, ENCD_, B_);
    gemm(stream, pcat, HD_ + ENCD_, projw, projb,
         outp + (size_t)t * VT_, (long long)(T_ - 1) * VT_, nullptr, 0,
         B_, VT_, HD_ + ENCD_, 0, 4);
  }
}
